// TeacherModel_12197707120732
// MI455X (gfx1250) — compile-verified, matrix operand order checked by reading, not run on hardware
//
#include <hip/hip_runtime.h>
#include <hip/hip_bf16.h>
#include <math.h>

// ---------------- model dimensions ----------------
#define BT     2
#define EDUN   49
#define TT     50
#define NN     50
#define HSZ    512
#define NHEADS 8
#define DHD    64
#define HHS    256
#define G3     768          // 3*HHS
#define PD     128
#define NTY    17
#define GLV    300
#define NE     4950          // nonzeros of _zm per batch = 50*(2*50-1)
#define ROWS_W (BT*EDUN)     // 98
#define BN     (BT*NN)       // 100
#define RB_ROWS (BN*NN)      // 5000 rows of nodes_bn
#define PN_ROWS (BN*NN*NN)   // 250000 path rows

typedef _Float16 v16h __attribute__((ext_vector_type(16)));
typedef float    v8f  __attribute__((ext_vector_type(8)));

// ---------------- WMMA fragment helpers ----------------
// Fragment layout (16-bit A/B, 16x16x32): lane r in [0,16): K = kb..kb+7 and
// kb+16..kb+23 where kb = k0 (+8 for lanes 16..31). Loads are 2x float4 pairs.
__device__ __forceinline__ v16h frag_ld(const float* __restrict__ p) {
    const float4* q = (const float4*)p;       // p is 16B aligned (lda/ldw multiples of 4)
    float4 x0 = q[0], x1 = q[1], x2 = q[4], x3 = q[5];
    v16h f;
    f[0]=(_Float16)x0.x;  f[1]=(_Float16)x0.y;  f[2]=(_Float16)x0.z;  f[3]=(_Float16)x0.w;
    f[4]=(_Float16)x1.x;  f[5]=(_Float16)x1.y;  f[6]=(_Float16)x1.z;  f[7]=(_Float16)x1.w;
    f[8]=(_Float16)x2.x;  f[9]=(_Float16)x2.y;  f[10]=(_Float16)x2.z; f[11]=(_Float16)x2.w;
    f[12]=(_Float16)x3.x; f[13]=(_Float16)x3.y; f[14]=(_Float16)x3.z; f[15]=(_Float16)x3.w;
    return f;
}

// K-tail fragment (only the K=300 GEMM takes this path; runs once per wave)
__device__ __forceinline__ v16h frag_ld_tail(const float* __restrict__ row, int kb, int K) {
    v16h f;
#pragma unroll
    for (int i = 0; i < 16; ++i) f[i] = (_Float16)0.f;
#pragma unroll
    for (int i = 0; i < 8; ++i) {
        int k1 = kb + i, k2 = kb + 16 + i;
        if (k1 < K) f[i]     = (_Float16)row[k1];
        if (k2 < K) f[i + 8] = (_Float16)row[k2];
    }
    return f;
}

__device__ __forceinline__ void store_tile(float* __restrict__ C, int ldc,
                                           const float* __restrict__ bias, v8f acc,
                                           int mbase, int nbase, int r, int hs,
                                           int M, int N, int act, int accum)
{
    int n = nbase + r;
    if (n >= N) return;                       // after all WMMAs; divergence is fine here
    float bv = bias ? bias[n] : 0.f;
#pragma unroll
    for (int vr = 0; vr < 8; ++vr) {
        int m = mbase + vr + hs * 8;
        if (m < M) {
            float v = acc[vr] + bv;
            if (act == 1)      v = fmaxf(v, 0.f);
            else if (act == 2) v = 1.f / (1.f + __expf(-v));
            else if (act == 3) v = tanhf(v);
            float* cp = &C[(size_t)m * ldc + n];
            if (accum) v += *cp;
            *cp = v;
        }
    }
}

// ---------------- generic WMMA GEMM ----------------
// C[m,n] (+)= act( sum_k A[m,k]*W[n,k] + bias[n] )
// A: MxK f32 (lda), W: NxK f32 (ldw), C: MxN f32 (ldc). lda/ldw multiples of 4.
// 32x32 output tile per wave (2x2 WMMA tiles, 4 accumulators).
// Out-of-range rows are CLAMPED, not guarded: rows m>=M / n>=N only feed
// outputs that are never stored, so garbage there is harmless -> branch-free
// inner loop of 16x global_load_b128 + cvt + 4x v_wmma.
__global__ void k_gemm(const float* __restrict__ A, int lda,
                       const float* __restrict__ W, int ldw,
                       const float* __restrict__ bias,
                       float* __restrict__ C, int ldc,
                       int M, int N, int K, int act, int accum)
{
    int wave = (int)((blockIdx.x * blockDim.x + threadIdx.x) >> 5); // wave-uniform
    int lane = threadIdx.x & 31;
    int mt = (M + 31) >> 5;
    int nt = (N + 31) >> 5;
    if (wave >= mt * nt) return;              // uniform exit: EXEC all-ones for WMMA
    int m0 = (wave / nt) << 5;
    int n0 = (wave % nt) << 5;
    int r  = lane & 15;
    int hs = lane >> 4;

    int am0 = m0 + r;        if (am0 > M - 1) am0 = M - 1;
    int am1 = m0 + 16 + r;   if (am1 > M - 1) am1 = M - 1;
    int bn0 = n0 + r;        if (bn0 > N - 1) bn0 = N - 1;
    int bn1 = n0 + 16 + r;   if (bn1 > N - 1) bn1 = N - 1;
    const float* Ar0 = A + (size_t)am0 * lda;
    const float* Ar1 = A + (size_t)am1 * lda;
    const float* Wr0 = W + (size_t)bn0 * ldw;
    const float* Wr1 = W + (size_t)bn1 * ldw;
    int ho = hs * 8;

    v8f acc00 = {0.f,0.f,0.f,0.f,0.f,0.f,0.f,0.f};
    v8f acc01 = acc00, acc10 = acc00, acc11 = acc00;

    int kfull = K & ~31;
    for (int k0 = 0; k0 < kfull; k0 += 32) {
        __builtin_prefetch(Ar0 + k0 + 256, 0, 0);   // global_prefetch_b8, ~8 steps ahead
        __builtin_prefetch(Wr0 + k0 + 256, 0, 0);
        v16h a0 = frag_ld(Ar0 + k0 + ho);
        v16h a1 = frag_ld(Ar1 + k0 + ho);
        v16h b0 = frag_ld(Wr0 + k0 + ho);
        v16h b1 = frag_ld(Wr1 + k0 + ho);
        acc00 = __builtin_amdgcn_wmma_f32_16x16x32_f16(false, a0, false, b0, (short)0, acc00, false, false);
        acc01 = __builtin_amdgcn_wmma_f32_16x16x32_f16(false, a0, false, b1, (short)0, acc01, false, false);
        acc10 = __builtin_amdgcn_wmma_f32_16x16x32_f16(false, a1, false, b0, (short)0, acc10, false, false);
        acc11 = __builtin_amdgcn_wmma_f32_16x16x32_f16(false, a1, false, b1, (short)0, acc11, false, false);
    }
    if (kfull < K) {                          // cold K-tail (K=300 case only)
        int kb = kfull + ho;
        v16h a0 = frag_ld_tail(Ar0, kb, K);
        v16h a1 = frag_ld_tail(Ar1, kb, K);
        v16h b0 = frag_ld_tail(Wr0, kb, K);
        v16h b1 = frag_ld_tail(Wr1, kb, K);
        acc00 = __builtin_amdgcn_wmma_f32_16x16x32_f16(false, a0, false, b0, (short)0, acc00, false, false);
        acc01 = __builtin_amdgcn_wmma_f32_16x16x32_f16(false, a0, false, b1, (short)0, acc01, false, false);
        acc10 = __builtin_amdgcn_wmma_f32_16x16x32_f16(false, a1, false, b0, (short)0, acc10, false, false);
        acc11 = __builtin_amdgcn_wmma_f32_16x16x32_f16(false, a1, false, b1, (short)0, acc11, false, false);
    }

    store_tile(C, ldc, bias, acc00, m0,      n0,      r, hs, M, N, act, accum);
    store_tile(C, ldc, bias, acc01, m0,      n0 + 16, r, hs, M, N, act, accum);
    store_tile(C, ldc, bias, acc10, m0 + 16, n0,      r, hs, M, N, act, accum);
    store_tile(C, ldc, bias, acc11, m0 + 16, n0 + 16, r, hs, M, N, act, accum);
}

// ---------------- elementwise / gather kernels ----------------
__global__ void k_fill(float* __restrict__ p, long n, float v) {
    long i = (long)blockIdx.x * blockDim.x + threadIdx.x;
    if (i < n) p[i] = v;
}

__global__ void k_embed(const int* __restrict__ texts, const float* __restrict__ emb,
                        float* __restrict__ out) {
    long i = (long)blockIdx.x * blockDim.x + threadIdx.x;
    if (i >= (long)ROWS_W * TT * GLV) return;
    int col = (int)(i % GLV);
    int row = (int)(i / GLV);
    int tok = texts[row];
    out[i] = emb[(size_t)tok * GLV + col];
}

// fused GRU gate: r=sig(ir+hr) z=sig(iz+hz) n=tanh(in+r*hn); h=(1-z)n+z h
__global__ void k_gru_gate(const float* __restrict__ xp_t, int xp_rs,
                           const float* __restrict__ gh,
                           float* __restrict__ h,
                           float* __restrict__ ys_t, int ys_rs, int rows)
{
    int idx = blockIdx.x * blockDim.x + threadIdx.x;
    if (idx >= rows * HHS) return;
    int row = idx / HHS, j = idx % HHS;
    const float* xt = xp_t + (size_t)row * xp_rs;
    const float* g  = gh   + (size_t)row * G3;
    float ir = xt[j], iz = xt[HHS + j], in = xt[2*HHS + j];
    float hr = g[j],  hz = g[HHS + j],  hn = g[2*HHS + j];
    float r = 1.f / (1.f + __expf(-(ir + hr)));
    float z = 1.f / (1.f + __expf(-(iz + hz)));
    float n = tanhf(in + r * hn);
    float hp = h[(size_t)row * HHS + j];
    float hn2 = (1.f - z) * n + z * hp;
    h[(size_t)row * HHS + j] = hn2;
    ys_t[(size_t)row * ys_rs + j] = hn2;
}

__global__ void k_gather_states(const float* __restrict__ ysf, const float* __restrict__ ysb,
                                const int* __restrict__ lengths, float* __restrict__ dlg)
{
    int idx = blockIdx.x * blockDim.x + threadIdx.x;
    if (idx >= BT * NN * HSZ) return;
    int h = idx % HSZ; int t = idx / HSZ; int n = t % NN; int b = t / NN;
    float v = 0.f;
    if (n > 0) {
        int be = b * EDUN + (n - 1);
        if (h < HHS) {
            int L = lengths[be];
            v = ysf[((size_t)be * TT + (L - 1)) * HHS + h];
        } else {
            v = ysb[((size_t)be * TT + 0) * HHS + (h - HHS)];
        }
    }
    dlg[idx] = v;
}

__global__ void k_concat_dirs(const float* __restrict__ f, const float* __restrict__ b,
                              float* __restrict__ out) {
    int idx = blockIdx.x * blockDim.x + threadIdx.x;
    if (idx >= BT * NN * HSZ) return;
    int c = idx % HSZ; int r = idx / HSZ;
    out[idx] = (c < HHS) ? f[(size_t)r * HHS + c] : b[(size_t)r * HHS + (c - HHS)];
}

// wave-per-row layernorm with optional residual; safe when out==a or out==res
__global__ void k_layernorm(const float* __restrict__ a, const float* __restrict__ res,
                            const float* __restrict__ g, const float* __restrict__ be,
                            float* __restrict__ out, int rows, int dim)
{
    int wrow = (int)((blockIdx.x * blockDim.x + threadIdx.x) >> 5);
    int lane = threadIdx.x & 31;
    if (wrow >= rows) return;
    const float* x = a + (size_t)wrow * dim;
    const float* r = res ? res + (size_t)wrow * dim : nullptr;
    float s = 0.f, s2 = 0.f;
    for (int i = lane; i < dim; i += 32) {
        float v = x[i] + (r ? r[i] : 0.f);
        s += v; s2 += v * v;
    }
#pragma unroll
    for (int m = 16; m; m >>= 1) { s += __shfl_xor(s, m, 32); s2 += __shfl_xor(s2, m, 32); }
    float mean = s / dim;
    float var  = s2 / dim - mean * mean;
    float rstd = rsqrtf(var + 1e-5f);
    for (int i = lane; i < dim; i += 32) {
        float v = x[i] + (r ? r[i] : 0.f);
        out[(size_t)wrow * dim + i] = (v - mean) * rstd * g[i] + be[i];
    }
}

__global__ void k_broadcast_nodes(const float* __restrict__ nodes, float* __restrict__ nbn) {
    long idx = (long)blockIdx.x * blockDim.x + threadIdx.x;
    if (idx >= (long)BT * NN * NN * HSZ) return;
    int h = (int)(idx % HSZ); long t = idx / HSZ;
    int j = (int)(t % NN); t /= NN;
    int b = (int)(t / NN);
    nbn[idx] = nodes[((size_t)b * NN + j) * HSZ + h];
}

__global__ void k_struct_init(const int* __restrict__ graphs, const float* __restrict__ ntype,
                              float* __restrict__ s5) {
    long idx = (long)blockIdx.x * blockDim.x + threadIdx.x;
    if (idx >= (long)BT * NN * NN * NN * PD) return;
    int d = (int)(idx % PD); long t = idx / PD;
    int q = (int)(t % NN); t /= NN;
    int p = (int)(t % NN); t /= NN;
    int c = (int)(t % NN); int b = (int)(t / NN);
    float v = 0.f;
    if (p != c && q != c) {
        int g = graphs[((size_t)b * NN + p) * NN + q] + graphs[((size_t)b * NN + q) * NN + p];
        if (g > 0) v = ntype[(size_t)(g - 1) * PD + d];
    }
    s5[idx] = v;
}

// fused path LayerNorm + sk/sv projections (saves materializing the 128MB pn tensor)
__global__ void k_path_sksv(const float* __restrict__ s5,
                            const float* __restrict__ pg, const float* __restrict__ pb,
                            const float* __restrict__ skW, const float* __restrict__ skb,
                            const float* __restrict__ svW, const float* __restrict__ svb,
                            float* __restrict__ sk, float* __restrict__ sv, int rows)
{
    int wrow = (int)((blockIdx.x * blockDim.x + threadIdx.x) >> 5);
    int lane = threadIdx.x & 31;
    if (wrow >= rows) return;
    const float* x = s5 + (size_t)wrow * PD;
    float s = 0.f, s2 = 0.f;
    for (int i = lane; i < PD; i += 32) { float v = x[i]; s += v; s2 += v * v; }
#pragma unroll
    for (int m = 16; m; m >>= 1) { s += __shfl_xor(s, m, 32); s2 += __shfl_xor(s2, m, 32); }
    float mean = s * (1.f / PD);
    float var  = s2 * (1.f / PD) - mean * mean;
    float rstd = rsqrtf(var + 1e-5f);
    int o0 = lane, o1 = lane + 32;
    const float4* x4   = (const float4*)x;
    const float4* pg4  = (const float4*)pg;
    const float4* pb4  = (const float4*)pb;
    const float4* skW0 = (const float4*)(skW + (size_t)o0 * PD);
    const float4* skW1 = (const float4*)(skW + (size_t)o1 * PD);
    const float4* svW0 = (const float4*)(svW + (size_t)o0 * PD);
    const float4* svW1 = (const float4*)(svW + (size_t)o1 * PD);
    float a0 = 0.f, a1 = 0.f, b0 = 0.f, b1 = 0.f;
#pragma unroll 4
    for (int i = 0; i < PD / 4; ++i) {
        float4 xv = x4[i], gv = pg4[i], bv = pb4[i];
        float4 w0 = skW0[i], w1 = skW1[i], u0 = svW0[i], u1 = svW1[i];
        float n0 = (xv.x - mean) * rstd * gv.x + bv.x;
        float n1 = (xv.y - mean) * rstd * gv.y + bv.y;
        float n2 = (xv.z - mean) * rstd * gv.z + bv.z;
        float n3 = (xv.w - mean) * rstd * gv.w + bv.w;
        a0 += n0*w0.x + n1*w0.y + n2*w0.z + n3*w0.w;
        a1 += n0*w1.x + n1*w1.y + n2*w1.z + n3*w1.w;
        b0 += n0*u0.x + n1*u0.y + n2*u0.z + n3*u0.w;
        b1 += n0*u1.x + n1*u1.y + n2*u1.z + n3*u1.w;
    }
    size_t rb = (size_t)wrow * DHD;
    sk[rb + o0] = a0 + skb[o0];  sk[rb + o1] = a1 + skb[o1];
    sv[rb + o0] = b0 + svb[o0];  sv[rb + o1] = b1 + svb[o1];
}

// scores: w[rb,h,i,j] = (q . (k + sk)) * DH^-0.5 + mask
__global__ void k_score(const float* __restrict__ q, const float* __restrict__ k,
                        const float* __restrict__ sk, const int* __restrict__ edu,
                        float* __restrict__ w)
{
    long idx = (long)blockIdx.x * blockDim.x + threadIdx.x;
    if (idx >= (long)BN * NHEADS * NN * NN) return;
    int j = (int)(idx % NN); long t = idx / NN;
    int i = (int)(t % NN); t /= NN;
    int h = (int)(t % NHEADS); int rb = (int)(t / NHEADS);
    int b = rb / NN;
    const float4* qp = (const float4*)(q  + ((size_t)rb * NN + i) * HSZ + h * DHD);
    const float4* kp = (const float4*)(k  + ((size_t)rb * NN + j) * HSZ + h * DHD);
    const float4* sp = (const float4*)(sk + (((size_t)rb * NN + i) * NN + j) * DHD);
    float s = 0.f;
#pragma unroll 4
    for (int d = 0; d < DHD / 4; ++d) {
        float4 qv = qp[d], kv = kp[d], sv = sp[d];
        s += qv.x * (kv.x + sv.x) + qv.y * (kv.y + sv.y)
           + qv.z * (kv.z + sv.z) + qv.w * (kv.w + sv.w);
    }
    s *= 0.125f;                       // DH^{-1/2} = 1/8
    if (j >= edu[b] + 1) s += -1e9f;
    w[idx] = s;
}

__global__ void k_softmax50(float* __restrict__ w, int rows) {
    int r = blockIdx.x * blockDim.x + threadIdx.x;
    if (r >= rows) return;
    float* p = w + (size_t)r * NN;
    float mx = -1e30f;
    for (int j = 0; j < NN; ++j) mx = fmaxf(mx, p[j]);
    float s = 0.f;
    for (int j = 0; j < NN; ++j) { float e = __expf(p[j] - mx); p[j] = e; s += e; }
    float inv = 1.f / s;
    for (int j = 0; j < NN; ++j) p[j] *= inv;
}

// out[rb,i,h*64+d] = sum_j w[rb,h,i,j] * (v[rb,j,h*64+d] + sv[rb,i,j,d])
__global__ void k_attnout(const float* __restrict__ w, const float* __restrict__ v,
                          const float* __restrict__ sv, float* __restrict__ out)
{
    long idx = (long)blockIdx.x * blockDim.x + threadIdx.x;
    if (idx >= (long)RB_ROWS * HSZ) return;
    int hd = (int)(idx % HSZ); long t = idx / HSZ;
    int i = (int)(t % NN); int rb = (int)(t / NN);
    int h = hd / DHD, d = hd % DHD;
    const float* wp = w + (((size_t)rb * NHEADS + h) * NN + i) * NN;
    float acc = 0.f;
    for (int j = 0; j < NN; ++j) {
        float vv = v[((size_t)rb * NN + j) * HSZ + hd] +
                   sv[(((size_t)rb * NN + i) * NN + j) * DHD + d];
        acc += wp[j] * vv;
    }
    out[idx] = acc;
}

// decode nonzero index of _zm (row-major nonzero order of (c,p,q))
__device__ __forceinline__ void decode_cpq(int e, int& c, int& p, int& q) {
    c = e / (2 * NN - 1);
    int o = e - c * (2 * NN - 1);
    if (o < c)           { p = o;          q = c;     }
    else if (o < c + NN) { p = c;          q = o - c; }
    else                 { p = o - NN + 1; q = c;     }
}

__global__ void k_gather_x(const float* __restrict__ nbn, float* __restrict__ x) {
    long idx = (long)blockIdx.x * blockDim.x + threadIdx.x;
    if (idx >= (long)BT * NE * 2 * HSZ) return;
    int col = (int)(idx % (2 * HSZ)); long t = idx / (2 * HSZ);
    int e = (int)(t % NE); int b = (int)(t / NE);
    int c, p, q; decode_cpq(e, c, p, q);
    int node = (col < HSZ) ? q : p;
    int h = col & (HSZ - 1);
    x[idx] = nbn[(((size_t)b * NN + c) * NN + node) * HSZ + h];
}

__global__ void k_pu_gate(const float* __restrict__ rt, const float* __restrict__ zt,
                          const float* __restrict__ ct, const float* __restrict__ ut,
                          float* __restrict__ hx, long n)
{
    long i = (long)blockIdx.x * blockDim.x + threadIdx.x;
    if (i >= n) return;
    float r = 1.f / (1.f + __expf(-rt[i]));
    float z = 1.f / (1.f + __expf(-zt[i]));
    float u = tanhf(ct[i] + r * ut[i]);
    hx[i] = z * hx[i] + (1.f - z) * u;
}

__global__ void k_scatter_hx(const float* __restrict__ hx, float* __restrict__ s5) {
    long idx = (long)blockIdx.x * blockDim.x + threadIdx.x;
    if (idx >= (long)BT * NE * PD) return;
    int d = (int)(idx % PD); long t = idx / PD;
    int e = (int)(t % NE); int b = (int)(t / NE);
    int c, p, q; decode_cpq(e, c, p, q);
    s5[((((size_t)b * NN + c) * NN + p) * NN + q) * PD + d] = hx[idx];
}

__global__ void k_pred(const float* __restrict__ s5, float* __restrict__ pred) {
    long idx = (long)blockIdx.x * blockDim.x + threadIdx.x;
    if (idx >= (long)BT * NN * NN * 2 * PD) return;
    int col = (int)(idx % (2 * PD)); long t = idx / (2 * PD);
    int j = (int)(t % NN); t /= NN;
    int i = (int)(t % NN); int b = (int)(t / NN);
    int d = col & (PD - 1);
    size_t src;
    if (col < PD) src = ((((size_t)b * NN + i) * NN + i) * NN + j) * PD + d;
    else          src = ((((size_t)b * NN + j) * NN + j) * NN + i) * PD + d;
    pred[idx] = s5[src];
}

// ---------------- host-side helpers ----------------
static inline int cdiv(long a, long b) { return (int)((a + b - 1) / b); }

static void gemm(hipStream_t s, const float* A, int lda, const float* W, int ldw,
                 const float* bias, float* C, int ldc, int M, int N, int K,
                 int act = 0, int accum = 0)
{
    int mt = (M + 31) >> 5, nt = (N + 31) >> 5;
    long waves = (long)mt * nt;
    k_gemm<<<cdiv(waves, 8), 256, 0, s>>>(A, lda, W, ldw, bias, C, ldc, M, N, K, act, accum);
}

static void run_gru_dir(hipStream_t s, const float* xp, const float* Wh, const float* bh,
                        float* h, float* gh, float* ys, int rows, bool reverse)
{
    for (int step = 0; step < TT; ++step) {
        int t = reverse ? (TT - 1 - step) : step;
        gemm(s, h, HHS, Wh, HHS, bh, gh, G3, rows, G3, HHS);
        int total = rows * HHS;
        k_gru_gate<<<cdiv(total, 256), 256, 0, s>>>(xp + (size_t)t * G3, TT * G3, gh, h,
                                                    ys + (size_t)t * HHS, TT * HHS, rows);
    }
}

// input index map (setup_inputs dict order; params dict flattened in insertion order)
enum {
    IN_TEXTS = 0, IN_LENGTHS, IN_EDU_NUMS, IN_SPEAKERS, IN_TURNS, IN_GRAPHS,
    P_EMB,
    P_SGRU_WI_F, P_SGRU_WH_F, P_SGRU_BI_F, P_SGRU_BH_F,
    P_SGRU_WI_B, P_SGRU_WH_B, P_SGRU_BI_B, P_SGRU_BH_B,
    P_DGRU_WI_F, P_DGRU_WH_F, P_DGRU_BI_F, P_DGRU_BH_F,
    P_DGRU_WI_B, P_DGRU_WH_B, P_DGRU_BI_B, P_DGRU_BH_B,
    P_NLN_G, P_NLN_B, P_SPK, P_TURN, P_POS, P_NORMAL_TYPE,
    P_PU_R_W, P_PU_R_B, P_PU_Z_W, P_PU_Z_B, P_PU_C_W, P_PU_C_B, P_PU_U_W, P_PU_U_B,
    P_Q_W, P_Q_B, P_K_W, P_K_B, P_V_W, P_V_B, P_O_W, P_O_B,
    P_SK_W, P_SK_B, P_SV_W, P_SV_B, P_PLN_G, P_PLN_B, P_LN_G, P_LN_B,
    P_LK1_W, P_LK1_B, P_LK2_W, P_LK2_B, P_LB1_W, P_LB1_B, P_LB2_W, P_LB2_B
};

extern "C" void kernel_launch(void* const* d_in, const int* in_sizes, int n_in,
                              void* d_out, int out_size, void* d_ws, size_t ws_size,
                              hipStream_t stream)
{
    (void)in_sizes; (void)n_in; (void)out_size; (void)ws_size;
    auto F = [&](int i) { return (const float*)d_in[i]; };
    auto Ii = [&](int i) { return (const int*)d_in[i]; };
    float* out = (float*)d_out;

    // bump allocator over d_ws (256B aligned slabs)
    char* base = (char*)d_ws;
    size_t off = 0;
    auto alloc = [&](size_t nfl) -> float* {
        float* p = (float*)(base + off);
        off += ((nfl * sizeof(float) + 255) / 256) * 256;
        return p;
    };

    float* embx   = alloc((size_t)ROWS_W * TT * GLV);     // word embeddings
    float* xpf    = alloc((size_t)ROWS_W * TT * G3);
    float* xpb    = alloc((size_t)ROWS_W * TT * G3);
    float* hbuf   = alloc((size_t)ROWS_W * HHS);
    float* ghbuf  = alloc((size_t)ROWS_W * G3);
    float* ysf    = alloc((size_t)ROWS_W * TT * HHS);
    float* ysb    = alloc((size_t)ROWS_W * TT * HHS);
    float* dlg_in = alloc((size_t)BT * NN * HSZ);
    float* dxpf   = alloc((size_t)BT * TT * G3);
    float* dxpb   = alloc((size_t)BT * TT * G3);
    float* dhb    = alloc((size_t)BT * HHS);
    float* dghb   = alloc((size_t)BT * G3);
    float* dysf   = alloc((size_t)BT * TT * HHS);
    float* dysb   = alloc((size_t)BT * TT * HHS);
    float* dlgout = alloc((size_t)BT * NN * HSZ);
    float* nodes  = alloc((size_t)BT * NN * HSZ);
    float* nbn    = alloc((size_t)RB_ROWS * HSZ);
    float* s5     = alloc((size_t)BT * NN * NN * NN * PD); // 32M floats / 128 MB (fits L2)
    float* skb    = alloc((size_t)PN_ROWS * DHD);
    float* svb    = alloc((size_t)PN_ROWS * DHD);
    float* qb     = alloc((size_t)RB_ROWS * HSZ);
    float* kb     = alloc((size_t)RB_ROWS * HSZ);
    float* vb     = alloc((size_t)RB_ROWS * HSZ);
    float* wb     = alloc((size_t)BN * NHEADS * NN * NN);
    float* aout   = alloc((size_t)RB_ROWS * HSZ);
    float* otmp   = alloc((size_t)RB_ROWS * HSZ);
    float* xpu    = alloc((size_t)BT * NE * 2 * HSZ);
    float* hx     = alloc((size_t)BT * NE * PD);
    float* rt     = alloc((size_t)BT * NE * PD);
    float* zt     = alloc((size_t)BT * NE * PD);
    float* ct     = alloc((size_t)BT * NE * PD);
    float* ut     = alloc((size_t)BT * NE * PD);
    float* pred   = alloc((size_t)BT * NN * NN * 2 * PD);
    float* t1     = alloc((size_t)BT * NN * NN * PD);

    // ---- 1) word embeddings + word-level BiGRU ----
    k_embed<<<cdiv((long)ROWS_W * TT * GLV, 256), 256, 0, stream>>>(Ii(IN_TEXTS), F(P_EMB), embx);
    gemm(stream, embx, GLV, F(P_SGRU_WI_F), GLV, F(P_SGRU_BI_F), xpf, G3, ROWS_W * TT, G3, GLV);
    gemm(stream, embx, GLV, F(P_SGRU_WI_B), GLV, F(P_SGRU_BI_B), xpb, G3, ROWS_W * TT, G3, GLV);
    k_fill<<<cdiv((long)ROWS_W * HHS, 256), 256, 0, stream>>>(hbuf, (long)ROWS_W * HHS, 0.f);
    run_gru_dir(stream, xpf, F(P_SGRU_WH_F), F(P_SGRU_BH_F), hbuf, ghbuf, ysf, ROWS_W, false);
    k_fill<<<cdiv((long)ROWS_W * HHS, 256), 256, 0, stream>>>(hbuf, (long)ROWS_W * HHS, 0.f);
    run_gru_dir(stream, xpb, F(P_SGRU_WH_B), F(P_SGRU_BH_B), hbuf, ghbuf, ysb, ROWS_W, true);

    // ---- 2) dialogue input + dialogue BiGRU ----
    k_gather_states<<<cdiv((long)BT * NN * HSZ, 256), 256, 0, stream>>>(ysf, ysb, Ii(IN_LENGTHS), dlg_in);
    gemm(stream, dlg_in, HSZ, F(P_DGRU_WI_F), HSZ, F(P_DGRU_BI_F), dxpf, G3, BT * TT, G3, HSZ);
    gemm(stream, dlg_in, HSZ, F(P_DGRU_WI_B), HSZ, F(P_DGRU_BI_B), dxpb, G3, BT * TT, G3, HSZ);
    k_fill<<<cdiv((long)BT * HHS, 256), 256, 0, stream>>>(dhb, (long)BT * HHS, 0.f);
    run_gru_dir(stream, dxpf, F(P_DGRU_WH_F), F(P_DGRU_BH_F), dhb, dghb, dysf, BT, false);
    k_fill<<<cdiv((long)BT * HHS, 256), 256, 0, stream>>>(dhb, (long)BT * HHS, 0.f);
    run_gru_dir(stream, dxpb, F(P_DGRU_WH_B), F(P_DGRU_BH_B), dhb, dghb, dysb, BT, true);
    k_concat_dirs<<<cdiv((long)BT * NN * HSZ, 256), 256, 0, stream>>>(dysf, dysb, dlgout);
    k_layernorm<<<cdiv((long)BT * NN, 8), 256, 0, stream>>>(dlgout, dlg_in, F(P_NLN_G), F(P_NLN_B),
                                                            nodes, BT * NN, HSZ);
    k_broadcast_nodes<<<cdiv((long)BT * NN * NN * HSZ, 256), 256, 0, stream>>>(nodes, nbn);

    // ---- 3) struct5 init + hx init ----
    k_struct_init<<<cdiv((long)BT * NN * NN * NN * PD, 256), 256, 0, stream>>>(Ii(IN_GRAPHS),
                                                                               F(P_NORMAL_TYPE), s5);
    k_fill<<<cdiv((long)BT * NE * PD, 256), 256, 0, stream>>>(hx, (long)BT * NE * PD, 0.f);

    // ---- 4) two layers of structure-aware attention + path GRU update ----
    for (int l = 0; l < 2; ++l) {
        gemm(stream, nbn, HSZ, F(P_Q_W), HSZ, F(P_Q_B), qb, HSZ, RB_ROWS, HSZ, HSZ);
        gemm(stream, nbn, HSZ, F(P_K_W), HSZ, F(P_K_B), kb, HSZ, RB_ROWS, HSZ, HSZ);
        gemm(stream, nbn, HSZ, F(P_V_W), HSZ, F(P_V_B), vb, HSZ, RB_ROWS, HSZ, HSZ);
        k_path_sksv<<<cdiv((long)PN_ROWS, 8), 256, 0, stream>>>(s5, F(P_PLN_G), F(P_PLN_B),
                                                                F(P_SK_W), F(P_SK_B),
                                                                F(P_SV_W), F(P_SV_B),
                                                                skb, svb, PN_ROWS);
        k_score<<<cdiv((long)BN * NHEADS * NN * NN, 256), 256, 0, stream>>>(qb, kb, skb,
                                                                            Ii(IN_EDU_NUMS), wb);
        k_softmax50<<<cdiv((long)BN * NHEADS * NN, 256), 256, 0, stream>>>(wb, BN * NHEADS * NN);
        k_attnout<<<cdiv((long)RB_ROWS * HSZ, 256), 256, 0, stream>>>(wb, vb, svb, aout);
        gemm(stream, aout, HSZ, F(P_O_W), HSZ, F(P_O_B), otmp, HSZ, RB_ROWS, HSZ, HSZ, /*relu*/1);
        k_layernorm<<<cdiv((long)RB_ROWS, 8), 256, 0, stream>>>(otmp, nbn, F(P_LN_G), F(P_LN_B),
                                                                nbn, RB_ROWS, HSZ);
        // path GRU
        k_gather_x<<<cdiv((long)BT * NE * 2 * HSZ, 256), 256, 0, stream>>>(nbn, xpu);
        gemm(stream, xpu, 2 * HSZ, F(P_PU_R_W), 2 * HSZ + PD, F(P_PU_R_B), rt, PD,
             BT * NE, PD, 2 * HSZ);
        gemm(stream, hx, PD, F(P_PU_R_W) + 2 * HSZ, 2 * HSZ + PD, nullptr, rt, PD,
             BT * NE, PD, PD, 0, /*accum*/1);
        gemm(stream, xpu, 2 * HSZ, F(P_PU_Z_W), 2 * HSZ + PD, F(P_PU_Z_B), zt, PD,
             BT * NE, PD, 2 * HSZ);
        gemm(stream, hx, PD, F(P_PU_Z_W) + 2 * HSZ, 2 * HSZ + PD, nullptr, zt, PD,
             BT * NE, PD, PD, 0, /*accum*/1);
        gemm(stream, xpu, 2 * HSZ, F(P_PU_C_W), 2 * HSZ, F(P_PU_C_B), ct, PD,
             BT * NE, PD, 2 * HSZ);
        gemm(stream, hx, PD, F(P_PU_U_W), PD, F(P_PU_U_B), ut, PD, BT * NE, PD, PD);
        k_pu_gate<<<cdiv((long)BT * NE * PD, 256), 256, 0, stream>>>(rt, zt, ct, ut, hx,
                                                                     (long)BT * NE * PD);
        k_scatter_hx<<<cdiv((long)BT * NE * PD, 256), 256, 0, stream>>>(hx, s5);
    }

    // ---- 5) link / label heads ----
    k_pred<<<cdiv((long)BT * NN * NN * 2 * PD, 256), 256, 0, stream>>>(s5, pred);
    // link = (tanh(pred@lk1^T+b) @ lk2^T + b)[...,0]   -> out[0 : 5000]
    gemm(stream, pred, 2 * PD, F(P_LK1_W), 2 * PD, F(P_LK1_B), t1, PD,
         BT * NN * NN, PD, 2 * PD, /*tanh*/3);
    gemm(stream, t1, PD, F(P_LK2_W), PD, F(P_LK2_B), out, 1, BT * NN * NN, 1, PD);
    // label = tanh(pred@lb1^T+b) @ lb2^T + b          -> out[5000 : 5000+85000]
    gemm(stream, pred, 2 * PD, F(P_LB1_W), 2 * PD, F(P_LB1_B), t1, PD,
         BT * NN * NN, PD, 2 * PD, /*tanh*/3);
    gemm(stream, t1, PD, F(P_LB2_W), PD, F(P_LB2_B), out + (size_t)BT * NN * NN, NTY,
         BT * NN * NN, NTY, PD);
}
